// GNN_TrackLinkingNet_22204980920777
// MI455X (gfx1250) — matrix operand order repaired and checked by reading, not compiled, not run on hardware
//
#include <hip/hip_runtime.h>

// Problem constants (fixed by the reference)
#define NN 100000   // nodes
#define NE 1600000  // edges
#define XD 19       // node input dim
#define EFD 12      // edge feature dim
#define HD 16       // hidden dim (= edge hidden dim)

typedef _Float16 v16h __attribute__((ext_vector_type(16)));
typedef float    v8f  __attribute__((ext_vector_type(8)));

#define DEV static __device__ __forceinline__

DEV float nan0(float v)  { return v == v ? v : 0.f; }
// slope 0.01 < 1  =>  leaky_relu(x) == max(x, 0.01x)  (v_mul + v_max, dual-issuable)
DEV float lrelu(float v) { return fmaxf(v, 0.01f * v); }
DEV float sigm(float v)  { return 1.f / (1.f + expf(-v)); }

DEV v8f wmma16(v16h a, v16h b, v8f c) {
  // D = A(16x32 f16) * B(32x16 f16) + C(16x16 f32)
  return __builtin_amdgcn_wmma_f32_16x16x32_f16(false, a, false, b, (short)0, c,
                                                false, false);
}

// ---------------------------------------------------------------- utilities
__global__ void fill0(float* __restrict__ p, int n) {
  int i = blockIdx.x * blockDim.x + threadIdx.x;
  if (i < n) p[i] = 0.f;
}

// Column mean/std statistics: register accum -> LDS atomics -> global atomics
template <int C>
__global__ void colstats(const float* __restrict__ data, long rows,
                         float* __restrict__ gsum, float* __restrict__ gsq) {
  __shared__ float ssum[C];
  __shared__ float ssq[C];
  for (int c = threadIdx.x; c < C; c += blockDim.x) { ssum[c] = 0.f; ssq[c] = 0.f; }
  __syncthreads();
  float ls[C], lq[C];
#pragma unroll
  for (int c = 0; c < C; ++c) { ls[c] = 0.f; lq[c] = 0.f; }
  const long stride = (long)gridDim.x * blockDim.x;
  for (long r = (long)blockIdx.x * blockDim.x + threadIdx.x; r < rows; r += stride) {
    const float* row = data + r * C;
#pragma unroll
    for (int c = 0; c < C; ++c) {
      float v = nan0(row[c]);
      ls[c] += v; lq[c] += v * v;
    }
  }
#pragma unroll
  for (int c = 0; c < C; ++c) { atomicAdd(&ssum[c], ls[c]); atomicAdd(&ssq[c], lq[c]); }
  __syncthreads();
  for (int c = threadIdx.x; c < C; c += blockDim.x) {
    atomicAdd(&gsum[c], ssum[c]); atomicAdd(&gsq[c], ssq[c]);
  }
}

__global__ void finalize_stats(const float* __restrict__ sum, const float* __restrict__ sq,
                               float* __restrict__ mean, float* __restrict__ inv,
                               int C, float rows) {
  int c = threadIdx.x;
  if (c < C) {
    float m = sum[c] / rows;
    float var = sq[c] / rows - m * m;
    var = var > 0.f ? var : 0.f;
    mean[c] = m;
    inv[c] = 1.f / (sqrtf(var) + 1e-4f);
  }
}

// ------------------------------------------------------- weight pre-packing
// B fragment layout (V_WMMA_*_16X16X32 B = 32x16, f16):
//   lane l: col = l%16, K = (l/16)*8 + (t%8) + (t/8)*16   (t = 0..15, 2 halves per VGPR)
// Pack each K x dout weight (K zero-padded to 32) so a lane reads 32B.
DEV void pack_frag(const float* __restrict__ W, int Krows, int dout, int colbase,
                   int krowbase, _Float16* __restrict__ dst, int tid, int nthr) {
  for (int idx = tid; idx < 512; idx += nthr) {
    int l = idx >> 4, t = idx & 15;
    int col = colbase + (l & 15);
    int k = krowbase + ((l >> 4) << 3) + (t & 7) + ((t >> 3) << 4);
    float v = 0.f;
    if (k < Krows && col < dout) v = W[k * dout + col];
    dst[idx] = (_Float16)v;
  }
}

__global__ void pack_weights_kernel(
    const float* wi1, const float* wi2, const float* we1, const float* we2,
    const float* watd, const float* watr, const float* c0l1, const float* c0l2,
    const float* c1l1, const float* c1l2, const float* en1, _Float16* pack) {
  int tid = threadIdx.x, nt = blockDim.x;
  pack_frag(wi1, 19, 16, 0, 0,  pack + 0  * 512, tid, nt);   // inp.W1
  pack_frag(wi2, 16, 16, 0, 0,  pack + 1  * 512, tid, nt);   // inp.W2
  pack_frag(we1, 12, 16, 0, 0,  pack + 2  * 512, tid, nt);   // edge_inp.W1
  pack_frag(we2, 16, 16, 0, 0,  pack + 3  * 512, tid, nt);   // edge_inp.W2
  pack_frag(watd, 16, 16, 0, 0, pack + 4  * 512, tid, nt);   // att_dir.W1
  pack_frag(watr, 16, 16, 0, 0, pack + 5  * 512, tid, nt);   // att_rev.W1
  pack_frag(c0l1, 32, 32, 0, 0, pack + 6  * 512, tid, nt);   // conv0.l1 cols 0-15
  pack_frag(c0l1, 32, 32, 16, 0, pack + 7 * 512, tid, nt);   // conv0.l1 cols 16-31
  pack_frag(c0l2, 32, 16, 0, 0, pack + 8  * 512, tid, nt);   // conv0.l2
  pack_frag(c1l1, 32, 32, 0, 0, pack + 9  * 512, tid, nt);   // conv1.l1 cols 0-15
  pack_frag(c1l1, 32, 32, 16, 0, pack + 10 * 512, tid, nt);  // conv1.l1 cols 16-31
  pack_frag(c1l2, 32, 16, 0, 0, pack + 11 * 512, tid, nt);   // conv1.l2
  pack_frag(en1, 60, 16, 0, 0,  pack + 12 * 512, tid, nt);   // edge_net.W1 rows 0-31
  pack_frag(en1, 60, 16, 0, 32, pack + 13 * 512, tid, nt);   // edge_net.W1 rows 32-59
}

// -------------------------------------------- node MLP: norm -> 19->16 -> 16->16
// A fragment layout (16x32 f16): lanes 0-15 row=lane K={0-7,16-23};
//                                lanes 16-31 row=lane-16 K={8-15,24-31}
// Persistent grid-stride waves; weights/biases/stats hoisted out of the loop.
__global__ void node_mlp_kernel(const float* __restrict__ x, const float* __restrict__ stats,
                                const _Float16* __restrict__ pack,
                                const float* __restrict__ b1, const float* __restrict__ b2,
                                float* __restrict__ emb) {
  __shared__ alignas(32) _Float16 stg[8][16 * 16];
  const int lane = threadIdx.x & 31, wv = threadIdx.x >> 5;
  const bool hi = lane >= 16;
  const int col = lane & 15;
  const int kb = hi ? 8 : 0;
  const int wid = blockIdx.x * 8 + wv;
  const int wstride = gridDim.x * 8;
  const int ntiles = NN / 16;

  // hoisted loop-invariants
  const v16h w1 = *(const v16h*)(pack + 0 * 512 + lane * 16);
  const v16h w2 = *(const v16h*)(pack + 1 * 512 + lane * 16);
  const float bias1 = b1[col];
  const float bias2 = b2[col];
  float m0[8], s0[8], m1[8], s1[8];
#pragma unroll
  for (int t = 0; t < 8; ++t) {
    int k = kb + t, k2 = 16 + kb + t;
    m0[t] = (k < XD) ? stats[64 + k] : 0.f;
    s0[t] = (k < XD) ? stats[96 + k] : 0.f;
    m1[t] = (k2 < XD) ? stats[64 + k2] : 0.f;
    s1[t] = (k2 < XD) ? stats[96 + k2] : 0.f;
  }

  for (int tile = wid; tile < ntiles; tile += wstride) {
    const int r = tile * 16 + col;
    v16h a;
#pragma unroll
    for (int t = 0; t < 8; ++t) {
      int k = kb + t, k2 = 16 + kb + t;
      a[t]     = (_Float16)((k < XD)  ? (nan0(x[(size_t)r * XD + k])  - m0[t]) * s0[t] : 0.f);
      a[8 + t] = (_Float16)((k2 < XD) ? (nan0(x[(size_t)r * XD + k2]) - m1[t]) * s1[t] : 0.f);
    }
    v8f z = {};
    v8f c1 = wmma16(a, w1, z);
#pragma unroll
    for (int v = 0; v < 8; ++v) {
      int row = v + (hi ? 8 : 0);
      stg[wv][row * 16 + col] = (_Float16)lrelu(c1[v] + bias1);
    }
    __builtin_amdgcn_wave_barrier();
    v16h a2;
#pragma unroll
    for (int t = 0; t < 8; ++t) {
      a2[t] = stg[wv][col * 16 + kb + t];   // K 0-15 split across half-waves
      a2[8 + t] = (_Float16)0.f;            // K 16-31 zero (weights padded too)
    }
    v8f c2 = wmma16(a2, w2, z);
#pragma unroll
    for (int v = 0; v < 8; ++v) {
      int row = v + (hi ? 8 : 0);
      emb[(size_t)(tile * 16 + row) * HD + col] = lrelu(c2[v] + bias2);
    }
    __builtin_amdgcn_wave_barrier();
  }
}

// --------------------- edge MLP + attention: norm -> 12->16 -> 16->16, 2 heads
__global__ void edge_mlp_kernel(
    const float* __restrict__ ef, const float* __restrict__ stats,
    const _Float16* __restrict__ pack,
    const float* __restrict__ eb1, const float* __restrict__ eb2,
    const float* __restrict__ ad_b1, const float* __restrict__ ad_w2, const float* __restrict__ ad_b2,
    const float* __restrict__ ar_b1, const float* __restrict__ ar_w2, const float* __restrict__ ar_b2,
    _Float16* __restrict__ efn16, _Float16* __restrict__ efnn16, float* __restrict__ alpha) {
  __shared__ alignas(32) _Float16 stg[8][16 * 16];
  __shared__ float dt[8][16 * 16];
  const int lane = threadIdx.x & 31, wv = threadIdx.x >> 5;
  const bool hi = lane >= 16;
  const int col = lane & 15;
  const int kb = hi ? 8 : 0;
  const int wid = blockIdx.x * 8 + wv;
  const int wstride = gridDim.x * 8;
  const int ntiles = NE / 16;

  // hoisted loop-invariants
  const v16h wl1 = *(const v16h*)(pack + 2 * 512 + lane * 16);
  const v16h wl2 = *(const v16h*)(pack + 3 * 512 + lane * 16);
  const v16h wad = *(const v16h*)(pack + 4 * 512 + lane * 16);
  const v16h war = *(const v16h*)(pack + 5 * 512 + lane * 16);
  const float bias1 = eb1[col];
  const float bias2 = eb2[col];
  const float adb1 = ad_b1[col];
  const float arb1 = ar_b1[col];
  const float adb2 = ad_b2[0];
  const float arb2 = ar_b2[0];
  float em[EFD], es[EFD];
#pragma unroll
  for (int c = 0; c < EFD; ++c) { em[c] = stats[192 + c]; es[c] = stats[224 + c]; }

  for (int tile = wid; tile < ntiles; tile += wstride) {
    const int e = tile * 16 + col;
    float nrm[EFD];
#pragma unroll
    for (int c = 0; c < EFD; ++c)
      nrm[c] = (nan0(ef[(size_t)e * EFD + c]) - em[c]) * es[c];
    if (!hi) {
#pragma unroll
      for (int c = 0; c < EFD; ++c) efn16[(size_t)e * EFD + c] = (_Float16)nrm[c];
    }
    v16h a;
#pragma unroll
    for (int t = 0; t < 8; ++t) {
      int k = kb + t;
      a[t] = (_Float16)(k < EFD ? nrm[k] : 0.f);
      a[8 + t] = (_Float16)0.f;
    }
    v8f z = {};
    v8f c1 = wmma16(a, wl1, z);
#pragma unroll
    for (int v = 0; v < 8; ++v) {
      int row = v + (hi ? 8 : 0);
      stg[wv][row * 16 + col] = (_Float16)lrelu(c1[v] + bias1);
    }
    __builtin_amdgcn_wave_barrier();
    v16h a2;
#pragma unroll
    for (int t = 0; t < 8; ++t) {
      a2[t] = stg[wv][col * 16 + kb + t];
      a2[8 + t] = (_Float16)0.f;
    }
    v8f c2 = wmma16(a2, wl2, z);
    __builtin_amdgcn_wave_barrier();
#pragma unroll
    for (int v = 0; v < 8; ++v) {
      int row = v + (hi ? 8 : 0);
      float val = lrelu(c2[v] + bias2);
      efnn16[(size_t)(tile * 16 + row) * HD + col] = (_Float16)val;
      stg[wv][row * 16 + col] = (_Float16)val;   // stage ef_nn for attention A
    }
    __builtin_amdgcn_wave_barrier();
    v16h a3;
#pragma unroll
    for (int t = 0; t < 8; ++t) {
      a3[t] = stg[wv][col * 16 + kb + t];
      a3[8 + t] = (_Float16)0.f;
    }
    // attention head: dir
    v8f cd = wmma16(a3, wad, z);
#pragma unroll
    for (int v = 0; v < 8; ++v) {
      int row = v + (hi ? 8 : 0);
      dt[wv][row * 16 + col] = lrelu(cd[v] + adb1);
    }
    __builtin_amdgcn_wave_barrier();
    if (!hi) {
      float s = adb2;
#pragma unroll
      for (int cc = 0; cc < 16; ++cc) s += dt[wv][col * 16 + cc] * ad_w2[cc];
      alpha[(size_t)tile * 16 + col] = sigm(s);
    }
    __builtin_amdgcn_wave_barrier();
    // attention head: rev
    v8f cr = wmma16(a3, war, z);
#pragma unroll
    for (int v = 0; v < 8; ++v) {
      int row = v + (hi ? 8 : 0);
      dt[wv][row * 16 + col] = lrelu(cr[v] + arb1);
    }
    __builtin_amdgcn_wave_barrier();
    if (!hi) {
      float s = arb2;
#pragma unroll
      for (int cc = 0; cc < 16; ++cc) s += dt[wv][col * 16 + cc] * ar_w2[cc];
      alpha[(size_t)NE + (size_t)tile * 16 + col] = sigm(s);
    }
    __builtin_amdgcn_wave_barrier();
  }
}

// ------------------- conv message+aggregate: [xi|xj-xi] -> 32->32 -> 32->16
__global__ void conv_kernel(const int* __restrict__ eix, const float* __restrict__ alpha,
                            const float* __restrict__ emb,
                            const _Float16* __restrict__ l1a, const _Float16* __restrict__ l1b,
                            const _Float16* __restrict__ l2w,
                            const float* __restrict__ b1, const float* __restrict__ b2,
                            float* __restrict__ num, float* __restrict__ den) {
  __shared__ alignas(32) _Float16 stg[8][16 * 32];
  __shared__ int sidx[8][16];
  __shared__ float sw[8][16];
  const int lane = threadIdx.x & 31, wv = threadIdx.x >> 5;
  const bool hi = lane >= 16;
  const int col = lane & 15;
  const int kb = hi ? 8 : 0;
  const int wid = blockIdx.x * 8 + wv;
  const int wstride = gridDim.x * 8;
  const int ntiles = (NN + 2 * NE) / 16;   // 206250

  // hoisted loop-invariants: 3 weight fragments + biases
  const v16h bA = *(const v16h*)(l1a + lane * 16);
  const v16h bB = *(const v16h*)(l1b + lane * 16);
  const v16h bC = *(const v16h*)(l2w + lane * 16);
  const float biasA = b1[col];
  const float biasB = b1[16 + col];
  const float bias2 = b2[col];

  for (int tile = wid; tile < ntiles; tile += wstride) {
    const long k = (long)tile * 16 + col;
    int i, j; float wt;
    if (k < NN) { i = (int)k; j = i; wt = 1.f; }                  // self-loop
    else if (k < NN + (long)NE) {
      int e = (int)(k - NN); i = eix[e]; j = eix[NE + e]; wt = alpha[e];
    } else {
      int e = (int)(k - NN - (long)NE); i = eix[NE + e]; j = eix[e];
      wt = alpha[(size_t)NE + e];
    }
    sidx[wv][col] = i; sw[wv][col] = wt;
    v16h a;
#pragma unroll
    for (int t = 0; t < 8; ++t) {
      float xi = emb[(size_t)i * HD + kb + t];
      float xj = emb[(size_t)j * HD + kb + t];
      a[t] = (_Float16)xi;                 // K 0-15  : xi
      a[8 + t] = (_Float16)(xj - xi);      // K 16-31 : xj - xi
    }
    v8f z = {};
    v8f cA = wmma16(a, bA, z);
    v8f cB = wmma16(a, bB, z);
#pragma unroll
    for (int v = 0; v < 8; ++v) {
      int row = v + (hi ? 8 : 0);
      stg[wv][row * 32 + col]      = (_Float16)lrelu(cA[v] + biasA);
      stg[wv][row * 32 + 16 + col] = (_Float16)lrelu(cB[v] + biasB);
    }
    __builtin_amdgcn_wave_barrier();
    v16h a2;
#pragma unroll
    for (int t = 0; t < 8; ++t) {
      a2[t]     = stg[wv][col * 32 + kb + t];        // K {0-7|8-15}
      a2[8 + t] = stg[wv][col * 32 + 16 + kb + t];   // K {16-23|24-31}
    }
    v8f cm = wmma16(a2, bC, z);
#pragma unroll
    for (int v = 0; v < 8; ++v) {
      int row = v + (hi ? 8 : 0);
      float m = lrelu(cm[v] + bias2);
      int tgt = sidx[wv][row];
      atomicAdd(&num[(size_t)tgt * HD + col], sw[wv][row] * m);
    }
    if (col == 0) {
#pragma unroll
      for (int v = 0; v < 8; ++v) {
        int row = v + (hi ? 8 : 0);
        atomicAdd(&den[sidx[wv][row]], sw[wv][row]);
      }
    }
    __builtin_amdgcn_wave_barrier();
  }
}

__global__ void divide_kernel(const float* __restrict__ num, const float* __restrict__ den,
                              float* __restrict__ emb) {
  int idx = blockIdx.x * blockDim.x + threadIdx.x;
  if (idx < NN * HD) emb[idx] = num[idx] / den[idx >> 4];
}

// ---------------- final head: [src|dst|ef_nn|ef_norm] (60, pad 64) -> 16 -> 1
__global__ void edge_out_kernel(const int* __restrict__ eix, const float* __restrict__ emb,
                                const _Float16* __restrict__ efnn16, const _Float16* __restrict__ efn16,
                                const _Float16* __restrict__ en1a, const _Float16* __restrict__ en1b,
                                const float* __restrict__ b1, const float* __restrict__ w2,
                                const float* __restrict__ b2, float* __restrict__ out) {
  __shared__ float dt[8][16 * 16];
  const int lane = threadIdx.x & 31, wv = threadIdx.x >> 5;
  const bool hi = lane >= 16;
  const int col = lane & 15;
  const int kb = hi ? 8 : 0;
  const int wid = blockIdx.x * 8 + wv;
  const int wstride = gridDim.x * 8;
  const int ntiles = NE / 16;

  // hoisted loop-invariants
  const v16h wA = *(const v16h*)(en1a + lane * 16);
  const v16h wB = *(const v16h*)(en1b + lane * 16);
  const float bias = b1[col];
  const float b2v = b2[0];

  for (int tile = wid; tile < ntiles; tile += wstride) {
    const int e = tile * 16 + col;
    const int is = eix[e], id = eix[NE + e];
    v16h a;
#pragma unroll
    for (int t = 0; t < 8; ++t) {
      a[t]     = (_Float16)emb[(size_t)is * HD + kb + t];   // K 0-15 : node_src
      a[8 + t] = (_Float16)emb[(size_t)id * HD + kb + t];   // K 16-31: node_dst
    }
    v8f c = {};
    c = wmma16(a, wA, c);
    v16h a1;
#pragma unroll
    for (int t = 0; t < 8; ++t) {
      a1[t] = efnn16[(size_t)e * HD + kb + t];              // K 32-47: ef_nn
      int kk = kb + t;
      a1[8 + t] = kk < EFD ? efn16[(size_t)e * EFD + kk]    // K 48-59: ef_norm
                           : (_Float16)0.f;
    }
    c = wmma16(a1, wB, c);                                  // accumulate
#pragma unroll
    for (int v = 0; v < 8; ++v) {
      int row = v + (hi ? 8 : 0);
      dt[wv][row * 16 + col] = lrelu(c[v] + bias);
    }
    __builtin_amdgcn_wave_barrier();
    if (!hi) {
      float s = b2v;
#pragma unroll
      for (int cc = 0; cc < 16; ++cc) s += dt[wv][col * 16 + cc] * w2[cc];
      out[(size_t)tile * 16 + col] = sigm(s);
    }
    __builtin_amdgcn_wave_barrier();
  }
}

// ------------------------------------------------------------------- launch
extern "C" void kernel_launch(void* const* d_in, const int* in_sizes, int n_in,
                              void* d_out, int out_size, void* d_ws, size_t ws_size,
                              hipStream_t stream) {
  (void)in_sizes; (void)n_in; (void)out_size; (void)ws_size;
  // Inputs in setup_inputs() insertion order; params traversed in dict/list order.
  const float* x      = (const float*)d_in[0];
  const int*   eix    = (const int*)  d_in[1];   // [2,E] row-major: src=eix[e], dst=eix[E+e]
  const float* ef     = (const float*)d_in[2];
  const float* inp_w1 = (const float*)d_in[3];
  const float* inp_b1 = (const float*)d_in[4];
  const float* inp_w2 = (const float*)d_in[5];
  const float* inp_b2 = (const float*)d_in[6];
  const float* ein_w1 = (const float*)d_in[7];
  const float* ein_b1 = (const float*)d_in[8];
  const float* ein_w2 = (const float*)d_in[9];
  const float* ein_b2 = (const float*)d_in[10];
  const float* ad_w1  = (const float*)d_in[11];
  const float* ad_b1  = (const float*)d_in[12];
  const float* ad_w2  = (const float*)d_in[13];
  const float* ad_b2  = (const float*)d_in[14];
  const float* ar_w1  = (const float*)d_in[15];
  const float* ar_b1  = (const float*)d_in[16];
  const float* ar_w2  = (const float*)d_in[17];
  const float* ar_b2  = (const float*)d_in[18];
  const float* c0l1w  = (const float*)d_in[19];
  const float* c0l1b  = (const float*)d_in[20];
  const float* c0l2w  = (const float*)d_in[21];
  const float* c0l2b  = (const float*)d_in[22];
  const float* c1l1w  = (const float*)d_in[23];
  const float* c1l1b  = (const float*)d_in[24];
  const float* c1l2w  = (const float*)d_in[25];
  const float* c1l2b  = (const float*)d_in[26];
  const float* en_w1  = (const float*)d_in[27];
  const float* en_b1  = (const float*)d_in[28];
  const float* en_w2  = (const float*)d_in[29];
  const float* en_b2  = (const float*)d_in[30];
  float* out = (float*)d_out;

  // Workspace layout (all fragment/row bases 32B-aligned)
  char* ws = (char*)d_ws;
  float*    stats  = (float*)ws;                      // 256 floats of stats
  _Float16* pack   = (_Float16*)(ws + 1024);          // 14 frags * 1KB
  const size_t offEfn16  = 17408;
  const size_t offEfnn16 = offEfn16  + (size_t)NE * EFD * 2;
  const size_t offAlpha  = offEfnn16 + (size_t)NE * HD * 2;
  const size_t offEmb    = offAlpha  + (size_t)2 * NE * 4;
  const size_t offNum    = offEmb    + (size_t)NN * HD * 4;
  const size_t offDen    = offNum    + (size_t)NN * HD * 4;
  _Float16* efn16  = (_Float16*)(ws + offEfn16);
  _Float16* efnn16 = (_Float16*)(ws + offEfnn16);
  float*    alpha  = (float*)(ws + offAlpha);
  float*    emb    = (float*)(ws + offEmb);
  float*    numb   = (float*)(ws + offNum);
  float*    denb   = (float*)(ws + offDen);

  // 1) column stats (zeroed each call -> deterministic)
  fill0<<<1, 256, 0, stream>>>(stats, 256);
  colstats<XD><<<400, 256, 0, stream>>>(x, (long)NN, stats + 0, stats + 32);
  colstats<EFD><<<2048, 256, 0, stream>>>(ef, (long)NE, stats + 128, stats + 160);
  finalize_stats<<<1, 32, 0, stream>>>(stats + 0, stats + 32, stats + 64, stats + 96,
                                       XD, (float)NN);
  finalize_stats<<<1, 32, 0, stream>>>(stats + 128, stats + 160, stats + 192, stats + 224,
                                       EFD, (float)NE);

  // 2) pack all WMMA B fragments (f16, zero-padded)
  pack_weights_kernel<<<1, 512, 0, stream>>>(inp_w1, inp_w2, ein_w1, ein_w2, ad_w1, ar_w1,
                                             c0l1w, c0l2w, c1l1w, c1l2w, en_w1, pack);

  // 3) node & edge embedding MLPs + attention (persistent grid-stride waves)
  node_mlp_kernel<<<256, 256, 0, stream>>>(x, stats, pack, inp_b1, inp_b2, emb);
  edge_mlp_kernel<<<1024, 256, 0, stream>>>(ef, stats, pack, ein_b1, ein_b2,
      ad_b1, ad_w2, ad_b2, ar_b1, ar_w2, ar_b2, efn16, efnn16, alpha);

  // 4) two conv iterations (message MLP via WMMA, weighted-mean via atomics)
  for (int it = 0; it < 2; ++it) {
    fill0<<<(NN * HD + 255) / 256, 256, 0, stream>>>(numb, NN * HD);
    fill0<<<(NN + 255) / 256, 256, 0, stream>>>(denb, NN);
    const _Float16* l1a = pack + (it ? 9 : 6) * 512;
    const _Float16* l1b = pack + (it ? 10 : 7) * 512;
    const _Float16* l2w = pack + (it ? 11 : 8) * 512;
    const float* b1 = it ? c1l1b : c0l1b;
    const float* b2 = it ? c1l2b : c0l2b;
    conv_kernel<<<1024, 256, 0, stream>>>(eix, alpha, emb, l1a, l1b, l2w, b1, b2,
                                          numb, denb);
    divide_kernel<<<(NN * HD + 255) / 256, 256, 0, stream>>>(numb, denb, emb);
  }

  // 5) final edge head
  edge_out_kernel<<<1024, 256, 0, stream>>>(eix, emb, efnn16, efn16,
      pack + 12 * 512, pack + 13 * 512, en_b1, en_w2, en_b2, out);
}
